// HardGAO_36996848287787
// MI455X (gfx1250) — matrix operand order repaired and checked by reading, not compiled, not run on hardware
//
#include <hip/hip_runtime.h>
#include <hip/hip_bf16.h>

typedef __attribute__((ext_vector_type(16))) _Float16 v16h;
typedef __attribute__((ext_vector_type(8)))  float    v8f;

#define F_IN 128
#define HFO  512          // H * FO
#define NOUT 1024         // fused: [h | res]
#define DNBR 32
#define KTOP 8
#define NEG_SLOPE 0.2f

// ---------------------------------------------------------------- kernel 1
// y[n] = |feat[n] . p| / ||p||   ;  gate[n] = sigmoid(y[n])
// one wave (32 lanes) per node, 8 nodes per 256-thread block
__global__ void hgao_y_kernel(const float* __restrict__ feat,
                              const float* __restrict__ p,
                              float* __restrict__ y,
                              float* __restrict__ gate, int n_nodes) {
    __shared__ float psum[8];
    __shared__ float pnorm_s;
    int tid = threadIdx.x;
    float v  = (tid < F_IN) ? p[tid] : 0.0f;
    float sq = v * v;
    #pragma unroll
    for (int o = 16; o; o >>= 1) sq += __shfl_xor(sq, o, 32);
    if ((tid & 31) == 0) psum[tid >> 5] = sq;
    __syncthreads();
    if (tid == 0) {
        float s = 0.f;
        for (int w = 0; w < 8; ++w) s += psum[w];
        pnorm_s = sqrtf(s);
    }
    __syncthreads();
    float pn   = pnorm_s;
    int  wave  = tid >> 5, lane = tid & 31;
    int  node  = blockIdx.x * 8 + wave;
    if (node < n_nodes) {
        const float* fr = feat + (size_t)node * F_IN;
        float acc = 0.f;
        #pragma unroll
        for (int i = 0; i < F_IN / 32; ++i) acc += fr[lane + 32 * i] * p[lane + 32 * i];
        #pragma unroll
        for (int o = 16; o; o >>= 1) acc += __shfl_xor(acc, o, 32);
        if (lane == 0) {
            float yy = fabsf(acc) / pn;
            y[node]    = yy;
            gate[node] = 1.0f / (1.0f + __expf(-yy));
        }
    }
}

// ---------------------------------------------------------------- kernel 2
// top-8 of 32 neighbor scores per node; fully unrolled, no dynamic indexing
__global__ void hgao_topk_kernel(const long long* __restrict__ nbrs,
                                 const float* __restrict__ y,
                                 int* __restrict__ src, int n_nodes) {
    int n = blockIdx.x * blockDim.x + threadIdx.x;
    if (n >= n_nodes) return;
    float sc[DNBR]; int id[DNBR];
    #pragma unroll
    for (int j = 0; j < DNBR; ++j) {
        long long s = nbrs[(size_t)n * DNBR + j];
        id[j] = (int)s;
        sc[j] = y[s];
    }
    unsigned chosen = 0;
    for (int t = 0; t < KTOP; ++t) {
        float best = -1e30f; int bid = 0; int bbit = 0;
        #pragma unroll
        for (int j = 0; j < DNBR; ++j) {
            bool ok = !((chosen >> j) & 1u) && (sc[j] > best);
            if (ok) { best = sc[j]; bid = id[j]; bbit = j; }
        }
        chosen |= 1u << bbit;
        src[(size_t)n * KTOP + t] = bid;
    }
}

// ---------------------------------------------------------------- kernel 3
// pack A panel: g = gate[n]*feat  ->  f16, WMMA 16x32 A-tile per-lane layout
// panel block (mt,kt): 512 halfs, lane-major (lane*16 + j contiguous)
__global__ void hgao_packA(const float* __restrict__ feat,
                           const float* __restrict__ gate,
                           _Float16* __restrict__ Ap, int n_nodes) {
    size_t idx = (size_t)blockIdx.x * 256 + threadIdx.x;   // n*128 + k
    if (idx >= (size_t)n_nodes * F_IN) return;
    int n = (int)(idx >> 7), k = (int)(idx & 127);
    float g = gate[n] * feat[idx];
    int kt = k >> 5, kk = k & 31;
    int hi, j;
    if      (kk <  8) { hi = 0; j = kk;      }
    else if (kk < 16) { hi = 1; j = kk -  8; }
    else if (kk < 24) { hi = 0; j = kk -  8; }
    else              { hi = 1; j = kk - 16; }
    int lane = (n & 15) + (hi << 4);
    size_t dst = ((size_t)(n >> 4) * 4 + kt) * 512 + lane * 16 + j;
    Ap[dst] = (_Float16)g;
}

// ---------------------------------------------------------------- kernel 4
// pack B panel: [W_fc | W_res] -> f16, WMMA 32x16 B-tile per-lane layout
__global__ void hgao_packW(const float* __restrict__ Wfc,
                           const float* __restrict__ Wres,
                           _Float16* __restrict__ Bp) {
    int idx = blockIdx.x * 256 + threadIdx.x;              // k*1024 + col
    if (idx >= F_IN * NOUT) return;
    int k = idx >> 10, col = idx & 1023;
    float w = (col < HFO) ? Wfc[k * HFO + col] : Wres[k * HFO + (col - HFO)];
    int nt = col >> 4, kt = k >> 5, kk = k & 31;
    int lane = (col & 15) + (((kk >> 3) & 1) << 4);
    int j = (kk & 7) + ((kk >> 4) << 3);
    size_t dst = ((size_t)nt * 4 + kt) * 512 + lane * 16 + j;
    Bp[dst] = (_Float16)w;
}

// ---------------------------------------------------------------- kernel 5
// fused GEMM: [N,128]f16 x [128,1024]f16 -> h16[N,512]f16 + out[N,512]f32(res)
// 256 threads = 8 waves; each wave owns TWO adjacent 16x16 N-tiles so every
// A fragment feeds 2 WMMAs (8 WMMAs / wave, 2 independent accumulator chains)
__global__ void __launch_bounds__(256)
hgao_gemm_kernel(const _Float16* __restrict__ Ap,
                 const _Float16* __restrict__ Bp,
                 _Float16* __restrict__ h16,
                 float* __restrict__ out, int n_nodes) {
    int wave = threadIdx.x >> 5, lane = threadIdx.x & 31;
    int mt  = blockIdx.x >> 2;                        // M tile (16 rows)
    int nt0 = (((blockIdx.x & 3) << 3) + wave) * 2;   // first of 2 N-tiles
    const v16h* Ab = (const v16h*)(Ap + (size_t)mt * 4 * 512);
    const v16h* B0 = (const v16h*)(Bp + (size_t)nt0 * 4 * 512);
    const v16h* B1 = (const v16h*)(Bp + (size_t)(nt0 + 1) * 4 * 512);
    v8f acc0 = {}, acc1 = {};
    #pragma unroll
    for (int kt = 0; kt < 4; ++kt) {
        v16h a  = Ab[kt * 32 + lane];
        v16h b0 = B0[kt * 32 + lane];
        v16h b1 = B1[kt * 32 + lane];
        acc0 = __builtin_amdgcn_wmma_f32_16x16x32_f16(false, a, false, b0,
                                                      (short)0, acc0, false, false);
        acc1 = __builtin_amdgcn_wmma_f32_16x16x32_f16(false, a, false, b1,
                                                      (short)0, acc1, false, false);
    }
    int  row0 = (lane >= 16) ? 8 : 0;
    int  col  = lane & 15;
    int  nb   = mt * 16 + row0;                       // first node this lane writes
    bool full = (mt * 16 + 15) < n_nodes;             // wave-uniform
    // nt0 is even -> both tiles on the same side of the h/res boundary (nt=32)
    if (nt0 < 32) {                                   // h part (f16, L2-friendly)
        _Float16* ha = h16 + (size_t)nb * HFO + nt0 * 16 + col;
        _Float16* hb = ha + 16;
        if (full) {
            #pragma unroll
            for (int r = 0; r < 8; ++r) {
                ha[(size_t)r * HFO] = (_Float16)acc0[r];
                hb[(size_t)r * HFO] = (_Float16)acc1[r];
            }
        } else {
            #pragma unroll
            for (int r = 0; r < 8; ++r)
                if (nb + r < n_nodes) {
                    ha[(size_t)r * HFO] = (_Float16)acc0[r];
                    hb[(size_t)r * HFO] = (_Float16)acc1[r];
                }
        }
    } else {                                          // residual straight to out
        float* oa = out + (size_t)nb * HFO + (nt0 - 32) * 16 + col;
        float* ob = oa + 16;
        if (full) {
            #pragma unroll
            for (int r = 0; r < 8; ++r) {
                oa[(size_t)r * HFO] = acc0[r];
                ob[(size_t)r * HFO] = acc1[r];
            }
        } else {
            #pragma unroll
            for (int r = 0; r < 8; ++r)
                if (nb + r < n_nodes) {
                    oa[(size_t)r * HFO] = acc0[r];
                    ob[(size_t)r * HFO] = acc1[r];
                }
        }
    }
}

// ---------------------------------------------------------------- kernel 6
// el/er attention logits per (node, head)
__global__ void hgao_eler_kernel(const _Float16* __restrict__ h16,
                                 const float* __restrict__ attn_l,
                                 const float* __restrict__ attn_r,
                                 float* __restrict__ el,
                                 float* __restrict__ er, int n_nodes) {
    int idx = blockIdx.x * 256 + threadIdx.x;      // n*8 + head
    if (idx >= n_nodes * 8) return;
    int hh = idx & 7;
    const _Float16* hr = h16 + (size_t)(idx >> 3) * HFO + hh * 64;
    const float* al = attn_l + hh * 64;
    const float* ar = attn_r + hh * 64;
    float sl = 0.f, sr = 0.f;
    #pragma unroll 8
    for (int f = 0; f < 64; ++f) {
        float v = (float)hr[f];
        sl += v * al[f];
        sr += v * ar[f];
    }
    el[idx] = sl;
    er[idx] = sr;
}

// ---------------------------------------------------------------- kernel 7
// softmax over k, gather-weighted aggregation, ELU, residual add
// one wave per node; lane covers 16 contiguous cols -> head = lane/4
__global__ void __launch_bounds__(256)
hgao_final_kernel(const _Float16* __restrict__ h16,
                  const int* __restrict__ src,
                  const float* __restrict__ el,
                  const float* __restrict__ er,
                  float* __restrict__ out, int n_nodes) {
    int wave = threadIdx.x >> 5, lane = threadIdx.x & 31;
    int n = blockIdx.x * 8 + wave;
    if (n >= n_nodes) return;
    int hh = lane >> 2;                    // this lane's head
    int s[KTOP];
    #pragma unroll
    for (int k = 0; k < KTOP; ++k) s[k] = src[(size_t)n * KTOP + k];
    float ern = er[(size_t)n * 8 + hh];
    float e[KTOP];
    float m = -1e30f;
    #pragma unroll
    for (int k = 0; k < KTOP; ++k) {
        float v = el[(size_t)s[k] * 8 + hh] + ern;
        v = (v > 0.f) ? v : NEG_SLOPE * v;          // leaky relu
        e[k] = v;
        m = fmaxf(m, v);
    }
    float sum = 0.f;
    #pragma unroll
    for (int k = 0; k < KTOP; ++k) { e[k] = __expf(e[k] - m); sum += e[k]; }
    float inv = 1.0f / sum;

    int c0 = lane * 16;                    // 32 lanes x 16 cols = 512
    float acc[16];
    #pragma unroll
    for (int f = 0; f < 16; ++f) acc[f] = 0.f;
    #pragma unroll
    for (int k = 0; k < KTOP; ++k) {
        float a = e[k] * inv;
        const _Float16* hr = h16 + (size_t)s[k] * HFO + c0;   // 32B -> 2x b128
        #pragma unroll
        for (int f = 0; f < 16; ++f) acc[f] += a * (float)hr[f];
    }
    float* orow = out + (size_t)n * HFO + c0;      // already holds residual
    #pragma unroll
    for (int f = 0; f < 16; ++f) {
        float v = acc[f];
        v = (v > 0.f) ? v : (__expf(v) - 1.0f);    // elu
        orow[f] = v + orow[f];
    }
}

// ---------------------------------------------------------------- launch
static inline size_t alignup(size_t x) { return (x + 255) & ~(size_t)255; }

extern "C" void kernel_launch(void* const* d_in, const int* in_sizes, int n_in,
                              void* d_out, int out_size, void* d_ws, size_t ws_size,
                              hipStream_t stream) {
    const float*     feat   = (const float*)d_in[0];
    const long long* nbrs   = (const long long*)d_in[1];
    const float*     p      = (const float*)d_in[2];
    const float*     Wfc    = (const float*)d_in[3];
    const float*     attn_l = (const float*)d_in[4];
    const float*     attn_r = (const float*)d_in[5];
    const float*     Wres   = (const float*)d_in[6];
    float*           out    = (float*)d_out;

    const int N  = in_sizes[0] / F_IN;             // 100000
    const int Mt = (N + 15) / 16;                  // 6250 M-tiles

    // workspace carve-up (~138 MB)
    char*  ws = (char*)d_ws;
    size_t o  = 0;
    float*    y    = (float*)(ws + o);    o += alignup((size_t)N * 4);
    float*    gate = (float*)(ws + o);    o += alignup((size_t)N * 4);
    int*      src  = (int*)(ws + o);      o += alignup((size_t)N * KTOP * 4);
    float*    el   = (float*)(ws + o);    o += alignup((size_t)N * 8 * 4);
    float*    er   = (float*)(ws + o);    o += alignup((size_t)N * 8 * 4);
    _Float16* Ap   = (_Float16*)(ws + o); o += alignup((size_t)Mt * 16 * F_IN * 2);
    _Float16* Bp   = (_Float16*)(ws + o); o += alignup((size_t)F_IN * NOUT * 2);
    _Float16* h16  = (_Float16*)(ws + o); o += alignup((size_t)Mt * 16 * HFO * 2);
    (void)ws_size;

    // 1. scores + gates
    hgao_y_kernel<<<(N + 7) / 8, 256, 0, stream>>>(feat, p, y, gate, N);
    // 2. top-k neighbor selection
    hgao_topk_kernel<<<(N + 255) / 256, 256, 0, stream>>>(nbrs, y, src, N);
    // 3. pack A (gated features, f16 WMMA layout)
    {
        size_t tot = (size_t)N * F_IN;
        hgao_packA<<<(int)((tot + 255) / 256), 256, 0, stream>>>(feat, gate, Ap, N);
    }
    // 4. pack B ([W_fc | W_res], f16 WMMA layout)
    hgao_packW<<<(F_IN * NOUT + 255) / 256, 256, 0, stream>>>(Wfc, Wres, Bp);
    // 5. fused WMMA GEMM -> h16 + residual into out (2 N-tiles per wave)
    hgao_gemm_kernel<<<Mt * 4, 256, 0, stream>>>(Ap, Bp, h16, out, N);
    // 6. attention logits
    hgao_eler_kernel<<<(N * 8 + 255) / 256, 256, 0, stream>>>(h16, attn_l, attn_r, el, er, N);
    // 7. softmax + aggregate + elu + residual
    hgao_final_kernel<<<(N + 7) / 8, 256, 0, stream>>>(h16, src, el, er, out, N);
}
